// VectorQuantizerEMA_56813827391972
// MI455X (gfx1250) — matrix operand order repaired and checked by reading, not compiled, not run on hardware
//
#include <hip/hip_runtime.h>
#include <hip/hip_bf16.h>

typedef __attribute__((ext_vector_type(16))) __bf16 v16bf;
typedef __attribute__((ext_vector_type(8)))  float  v8f;
typedef __attribute__((ext_vector_type(4)))  unsigned int v4u;
typedef __attribute__((ext_vector_type(8)))  int v8i;
typedef __attribute__((ext_vector_type(4)))  int v4i;

constexpr int DIM = 512;
constexpr int NE  = 4096;           // number of codes
constexpr int NROWS = 8 * 4096;     // B*S = 32768
constexpr int NSTAGE = (NE / 128) * (DIM / 32);   // 512 B-tile stages

#if __has_builtin(__builtin_amdgcn_tensor_load_to_lds)
#define USE_TDM 1
#else
#define USE_TDM 0
#endif

__device__ __forceinline__ unsigned short f2bf_rne(float f) {
  unsigned u = __float_as_uint(f);
  unsigned r = u + 0x7FFFu + ((u >> 16) & 1u);   // round-to-nearest-even bf16
  return (unsigned short)(r >> 16);
}
__device__ __forceinline__ float bf2f(unsigned short h) {
  return __uint_as_float(((unsigned)h) << 16);
}

union Frag {
  v16bf v;
  uint4 q[2];
  unsigned short s[16];
};

#if USE_TDM
// ---------------------------------------------------------------------------
// TDM: DMA a 128-row x 32-element (bf16) tile of a [4096 x 512] bf16 tensor
// into LDS. D# packed per CDNA5 ISA §8 (normal mode, 2D tile):
//   group0: count=1 | lds_addr | global_addr | type=2
//   group1: data_size=2B, tensor_dim0=512, tensor_dim1=4096, tile_dim0=32,
//           tile_dim1=128, tensor_dim0_stride=512, tensor_dim1_stride=512*4096
// LDS fill order = row-major 64B/row => identical to our tile layout.
// ---------------------------------------------------------------------------
__device__ __forceinline__ void tdm_load_tile(unsigned ldsOff,
                                              const unsigned short* gsrc) {
  unsigned long long ga = (unsigned long long)(uintptr_t)gsrc;
  v4u g0 = { 1u,                                   // count=1, user mode
             ldsOff,                               // lds_addr (bytes)
             (unsigned)(ga & 0xFFFFFFFFu),         // global_addr[31:0]
             ((unsigned)((ga >> 32) & 0x1FFFFFFu)) | 0x80000000u }; // [56:32] | type=2
  v8i g1 = { 0x10000,                              // data_size = 1 (2 bytes)
             (512 << 16),                          // tensor_dim0[15:0]
             (4096 << 16),                         // dim0 hi=0 | tensor_dim1[15:0]
             (32 << 16),                           // dim1 hi=0 | tile_dim0 = 32
             128,                                  // tile_dim1 = 128, tile_dim2 = 0
             512,                                  // tensor_dim0_stride[31:0]
             0,                                    // stride0 hi | stride1[15:0]=0
             32 };                                 // tensor_dim1_stride[47:16] = 2M>>16
  v4i g2 = { 1, 0, 0, 0 };                         // tensor_dim2=1, tile_dim3=0
  v4i g3 = { 0, 0, 0, 0 };
#if __has_include(<hip/amd_detail/amd_gfx1250_TDM.h>)
  v8i g4 = { 0, 0, 0, 0, 0, 0, 0, 0 };             // therock toolchain: 6-arg form
  __builtin_amdgcn_tensor_load_to_lds(g0, g1, g2, g3, g4, 0);
#else
  __builtin_amdgcn_tensor_load_to_lds(g0, g1, g2, g3, 0);  // ROCm 7.2: 5-arg form
#endif
}
#endif

// ---------------------------------------------------------------------------
// Prep A: split input rows into bf16 hi/lo once (removes all conversion VALU
// from the WMMA hot loop; re-reads then come from the 192MB L2).
// ---------------------------------------------------------------------------
__global__ void __launch_bounds__(256) vq_split(const float* __restrict__ in,
                                                unsigned short* __restrict__ xHi,
                                                unsigned short* __restrict__ xLo) {
  size_t i = ((size_t)blockIdx.x * 256 + threadIdx.x) * 4;
  float4 v = *reinterpret_cast<const float4*>(in + i);
  unsigned short h0 = f2bf_rne(v.x), h1 = f2bf_rne(v.y),
                 h2 = f2bf_rne(v.z), h3 = f2bf_rne(v.w);
  unsigned short l0 = f2bf_rne(v.x - bf2f(h0)), l1 = f2bf_rne(v.y - bf2f(h1)),
                 l2 = f2bf_rne(v.z - bf2f(h2)), l3 = f2bf_rne(v.w - bf2f(h3));
  uint2 hp, lp;
  hp.x = (unsigned)h0 | ((unsigned)h1 << 16);
  hp.y = (unsigned)h2 | ((unsigned)h3 << 16);
  lp.x = (unsigned)l0 | ((unsigned)l1 << 16);
  lp.y = (unsigned)l2 | ((unsigned)l3 << 16);
  *reinterpret_cast<uint2*>(xHi + i) = hp;
  *reinterpret_cast<uint2*>(xLo + i) = lp;
}

// ---------------------------------------------------------------------------
// Prep B: eT[f32, 4096x512], eHi/eLo[bf16, 4096x512] (transposed codebook),
//         norms[j] = ||e_j||^2
// ---------------------------------------------------------------------------
__global__ void __launch_bounds__(256) vq_prep(const float* __restrict__ embed,
                                               float* __restrict__ eT,
                                               unsigned short* __restrict__ eHi,
                                               unsigned short* __restrict__ eLo,
                                               float* __restrict__ norms) {
  __shared__ float tile[8][33];
  __shared__ float nsh[32][9];
  const int t  = threadIdx.x;
  const int j0 = blockIdx.x * 32;
  const int dl  = t >> 5, jl  = t & 31;   // load mapping (coalesced over j)
  const int jl2 = t >> 3, dl2 = t & 7;    // store mapping (contiguous over d)
  float nacc = 0.f;
  for (int d0 = 0; d0 < DIM; d0 += 8) {
    float v = embed[(size_t)(d0 + dl) * NE + (j0 + jl)];
    nacc += v * v;
    tile[dl][jl] = v;
    __syncthreads();
    float f = tile[dl2][jl2];
    size_t o = (size_t)(j0 + jl2) * DIM + (d0 + dl2);
    eT[o] = f;
    unsigned short h = f2bf_rne(f);
    eHi[o] = h;
    eLo[o] = f2bf_rne(f - bf2f(h));
    __syncthreads();
  }
  nsh[jl][dl] = nacc;
  __syncthreads();
  if (t < 32) {
    float s = 0.f;
    #pragma unroll
    for (int d = 0; d < 8; ++d) s += nsh[t][d];
    norms[j0 + t] = s;
  }
}

// ---------------------------------------------------------------------------
// Main: fused (||e||^2 - 2 x.e) + running argmin, bf16x3 split WMMA.
// Block = 256 threads = 8 waves; wave owns 16 rows; block owns 128 rows.
// B tiles DMAed into LDS by the Tensor Data Mover, double buffered.
// ---------------------------------------------------------------------------
__global__ void __launch_bounds__(256) vq_score(const unsigned short* __restrict__ xHi,
                                                const unsigned short* __restrict__ xLo,
                                                const unsigned short* __restrict__ eHi,
                                                const unsigned short* __restrict__ eLo,
                                                const float* __restrict__ norms,
                                                float* __restrict__ outIdxF,
                                                int* __restrict__ jstar) {
  // [buf][hi/lo][128 cols x 16 dwords]  (32 KB)
  __shared__ unsigned int tiles[2][2][128 * 16];

  const int tid    = threadIdx.x;
  const int wave   = tid >> 5;
  const int lane   = tid & 31;
  const int laneN  = lane & 15;
  const bool upper = lane >= 16;
  const int rowBase = blockIdx.x * 128 + wave * 16;

  float bestv[8];
  int   besti[8];
  #pragma unroll
  for (int r = 0; r < 8; ++r) { bestv[r] = 3.4e38f; besti[r] = 0; }

  const v8f vzero = {0.f, 0.f, 0.f, 0.f, 0.f, 0.f, 0.f, 0.f};

#if USE_TDM
  // NB: addrspace(3) offsets come from truncating the flat LDS pointer
  // (flat LDS aperture maps addr[31:0] to the LDS offset).
  const unsigned ldsHi0 = (unsigned)(uintptr_t)(void*)&tiles[0][0][0];
  const unsigned ldsLo0 = (unsigned)(uintptr_t)(void*)&tiles[0][1][0];
  const unsigned ldsHi1 = (unsigned)(uintptr_t)(void*)&tiles[1][0][0];
  const unsigned ldsLo1 = (unsigned)(uintptr_t)(void*)&tiles[1][1][0];
  if (wave == 0) {           // prologue: stage 0 -> buffer 0
    tdm_load_tile(ldsHi0, eHi);
    tdm_load_tile(ldsLo0, eLo);
  }
#endif

  for (int jb = 0; jb < NE; jb += 128) {
    float nrm[8];
    #pragma unroll
    for (int nt = 0; nt < 8; ++nt) nrm[nt] = norms[jb + nt * 16 + laneN];

    v8f acc[8];
    #pragma unroll
    for (int nt = 0; nt < 8; ++nt) acc[nt] = vzero;

    for (int kc = 0; kc < DIM / 32; ++kc) {
      const int s = (jb >> 7) * (DIM / 32) + kc;
#if USE_TDM
      __syncthreads();       // all waves done reading buffer (s+1)&1
      if (wave == 0) {
        const int sn = s + 1;
        if (sn < NSTAGE) {   // prefetch next stage into the other buffer
          const int jbn = (sn >> 4) << 7;
          const int kcn = sn & 15;
          const size_t off = (size_t)jbn * DIM + kcn * 32;
          tdm_load_tile((sn & 1) ? ldsHi1 : ldsHi0, eHi + off);
          tdm_load_tile((sn & 1) ? ldsLo1 : ldsLo0, eLo + off);
          __builtin_amdgcn_s_wait_tensorcnt(2);  // in-order: stage s complete
        } else {
          __builtin_amdgcn_s_wait_tensorcnt(0);
        }
      }
      __syncthreads();       // buffer s&1 now valid for everyone
      const unsigned int* tHi = tiles[s & 1][0];
      const unsigned int* tLo = tiles[s & 1][1];
#else
      __syncthreads();
      { // fallback: cooperative copy, single buffer
        int c = tid & 127;
        const uint4* src;
        uint4* dst;
        if (tid < 128) {
          src = reinterpret_cast<const uint4*>(eHi + (size_t)(jb + c) * DIM + kc * 32);
          dst = reinterpret_cast<uint4*>(&tiles[0][0][c * 16]);
        } else {
          src = reinterpret_cast<const uint4*>(eLo + (size_t)(jb + c) * DIM + kc * 32);
          dst = reinterpret_cast<uint4*>(&tiles[0][1][c * 16]);
        }
        dst[0] = src[0]; dst[1] = src[1]; dst[2] = src[2]; dst[3] = src[3];
      }
      __syncthreads();
      const unsigned int* tHi = tiles[0][0];
      const unsigned int* tLo = tiles[0][1];
#endif

      // A hi/lo fragments: pure b128 loads from the pre-split input (L2-hot).
      Frag aHi, aLo;
      {
        const size_t base = (size_t)(rowBase + laneN) * DIM + kc * 32 + (upper ? 8 : 0);
        const uint4* ph = reinterpret_cast<const uint4*>(xHi + base);
        const uint4* pl = reinterpret_cast<const uint4*>(xLo + base);
        aHi.q[0] = ph[0]; aHi.q[1] = ph[2];   // K runs {+0..7} and {+16..23}
        aLo.q[0] = pl[0]; aLo.q[1] = pl[2];
      }

      #pragma unroll
      for (int nt = 0; nt < 8; ++nt) {
        Frag bHi, bLo;
        const int c = nt * 16 + laneN;
        const uint4* ph = reinterpret_cast<const uint4*>(&tHi[c * 16 + (upper ? 4 : 0)]);
        const uint4* pl = reinterpret_cast<const uint4*>(&tLo[c * 16 + (upper ? 4 : 0)]);
        bHi.q[0] = ph[0]; bHi.q[1] = ph[2];
        bLo.q[0] = pl[0]; bLo.q[1] = pl[2];
        // D = A*B + C ; bf16x3 split: hi*hi + hi*lo + lo*hi  (~fp32 accurate)
        acc[nt] = __builtin_amdgcn_wmma_f32_16x16x32_bf16(false, aHi.v, false, bHi.v,
                                                          (short)0, acc[nt], false, false);
        acc[nt] = __builtin_amdgcn_wmma_f32_16x16x32_bf16(false, aHi.v, false, bLo.v,
                                                          (short)0, acc[nt], false, false);
        acc[nt] = __builtin_amdgcn_wmma_f32_16x16x32_bf16(false, aLo.v, false, bHi.v,
                                                          (short)0, acc[nt], false, false);
      }
    }

    // score = ||e||^2 - 2 x.e ; running per-lane argmin
    #pragma unroll
    for (int nt = 0; nt < 8; ++nt) {
      int col = jb + nt * 16 + laneN;
      #pragma unroll
      for (int r = 0; r < 8; ++r) {
        float score = nrm[nt] - 2.0f * acc[nt][r];
        if (score < bestv[r]) { bestv[r] = score; besti[r] = col; }
      }
    }
  }

  // Reduce across the 16 lanes that share each row (xor stays within half-wave).
  #pragma unroll
  for (int r = 0; r < 8; ++r) {
    #pragma unroll
    for (int off = 8; off >= 1; off >>= 1) {
      float ov = __shfl_xor(bestv[r], off, 32);
      int   oi = __shfl_xor(besti[r], off, 32);
      if (ov < bestv[r] || (ov == bestv[r] && oi < besti[r])) { bestv[r] = ov; besti[r] = oi; }
    }
  }
  if (laneN == 0) {
    #pragma unroll
    for (int r = 0; r < 8; ++r) {
      int row = rowBase + (upper ? 8 : 0) + r;   // C layout: VGPR r -> M=r / M=8+r
      jstar[row] = besti[r];
      outIdxF[row] = (float)besti[r];
    }
  }
}

// ---------------------------------------------------------------------------
// Gather quantize rows + per-row partial sum of (q - x)^2 (deterministic).
// ---------------------------------------------------------------------------
__global__ void __launch_bounds__(128) vq_gather(const float* __restrict__ input,
                                                 const float* __restrict__ eT,
                                                 const int* __restrict__ jstar,
                                                 float* __restrict__ quant,
                                                 float* __restrict__ partial) {
  __shared__ float red[128];
  const int row = blockIdx.x;
  const int t = threadIdx.x;
  const int j = jstar[row];
  const int d = t * 4;
  float4 q = *reinterpret_cast<const float4*>(eT + (size_t)j * DIM + d);
  float4 x = *reinterpret_cast<const float4*>(input + (size_t)row * DIM + d);
  *reinterpret_cast<float4*>(quant + (size_t)row * DIM + d) = q;
  float dx = q.x - x.x, dy = q.y - x.y, dz = q.z - x.z, dw = q.w - x.w;
  red[t] = dx * dx + dy * dy + dz * dz + dw * dw;
  __syncthreads();
  for (int off = 64; off >= 1; off >>= 1) {
    if (t < off) red[t] += red[t + off];
    __syncthreads();
  }
  if (t == 0) partial[row] = red[0];
}

__global__ void __launch_bounds__(256) vq_reduce(const float* __restrict__ partial,
                                                 float* __restrict__ diffOut) {
  __shared__ float red[256];
  const int t = threadIdx.x;
  float s = 0.f;
  for (int i = t; i < NROWS; i += 256) s += partial[i];
  red[t] = s;
  __syncthreads();
  for (int off = 128; off >= 1; off >>= 1) {
    if (t < off) red[t] += red[t + off];
    __syncthreads();
  }
  if (t == 0) *diffOut = red[0] * (1.0f / ((float)NROWS * (float)DIM));
}

// ---------------------------------------------------------------------------
extern "C" void kernel_launch(void* const* d_in, const int* in_sizes, int n_in,
                              void* d_out, int out_size, void* d_ws, size_t ws_size,
                              hipStream_t stream) {
  (void)in_sizes; (void)n_in; (void)out_size; (void)ws_size;
  const float* input = (const float*)d_in[0];   // [8,4096,512] f32
  const float* embed = (const float*)d_in[1];   // [512,4096]  f32
  float* out = (float*)d_out;

  // ws layout (bytes): eT 8MB | eHi 4MB | eLo 4MB | norms 16KB | partial 128KB
  //                    | jstar 128KB | xHi 32MB | xLo 32MB  (~80.3 MB total)
  char* ws = (char*)d_ws;
  float*          eT      = (float*)ws;
  unsigned short* eHi     = (unsigned short*)(ws + 8388608);
  unsigned short* eLo     = (unsigned short*)(ws + 12582912);
  float*          norms   = (float*)(ws + 16777216);
  float*          partial = (float*)(ws + 16793600);
  int*            jstar   = (int*)(ws + 16924672);
  unsigned short* xHi     = (unsigned short*)(ws + 17055744);
  unsigned short* xLo     = (unsigned short*)(ws + 50610176);

  float* quant   = out;                               // 16777216 f32
  float* diffOut = out + (size_t)NROWS * DIM;         // 1 f32
  float* idxOut  = diffOut + 1;                       // 32768 f32 (indices)

  vq_split <<<(NROWS * DIM) / 1024, 256, 0, stream>>>(input, xHi, xLo);
  vq_prep  <<<NE / 32,              256, 0, stream>>>(embed, eT, eHi, eLo, norms);
  vq_score <<<NROWS / 128,          256, 0, stream>>>(xHi, xLo, eHi, eLo, norms,
                                                      idxOut, jstar);
  vq_gather<<<NROWS,                128, 0, stream>>>(input, eT, jstar, quant, partial);
  vq_reduce<<<1,                    256, 0, stream>>>(partial, diffOut);
}
